// SuperEnhancedGraphClassifier_73555609911552
// MI455X (gfx1250) — matrix operand order repaired and checked by reading, not compiled, not run on hardware
//
#include <hip/hip_runtime.h>
#include <hip/hip_bf16.h>
#include <math.h>

typedef __attribute__((ext_vector_type(16))) _Float16 v16h;
typedef __attribute__((ext_vector_type(8)))  _Float16 v8h;
typedef __attribute__((ext_vector_type(8)))  float    v8f;

#define GEMM_KC 256          // K-chunk staged in LDS
#define GEMM_KS (GEMM_KC+8)  // padded column stride (halfs) -> conflict-free banks
#define GEMM_SHM (64 * GEMM_KS * 2)

// ---------- helpers ----------
__device__ __forceinline__ unsigned encF(float f) {
  unsigned b = __float_as_uint(f);
  return (b & 0x80000000u) ? ~b : (b | 0x80000000u);
}
__device__ __forceinline__ float decF(unsigned k) {
  return __uint_as_float((k & 0x80000000u) ? (k & 0x7fffffffu) : ~k);
}

__global__ void fill_f32(float* p, float v, size_t n) {
  size_t i = (size_t)blockIdx.x * blockDim.x + threadIdx.x;
  size_t st = (size_t)gridDim.x * blockDim.x;
  for (; i < n; i += st) p[i] = v;
}
__global__ void fill_u32(unsigned* p, unsigned v, size_t n) {
  size_t i = (size_t)blockIdx.x * blockDim.x + threadIdx.x;
  size_t st = (size_t)gridDim.x * blockDim.x;
  for (; i < n; i += st) p[i] = v;
}
__global__ void f2h_kernel(const float* __restrict__ a, _Float16* __restrict__ b, size_t n) {
  size_t i = (size_t)blockIdx.x * blockDim.x + threadIdx.x;
  size_t st = (size_t)gridDim.x * blockDim.x;
  for (; i < n; i += st) b[i] = (_Float16)a[i];
}
// W[K,Nc] (row-major f32) -> Wt[Nc,K] (row-major f16)
__global__ void f2h_tr_kernel(const float* __restrict__ W, _Float16* __restrict__ Wt,
                              int K, int Nc) {
  size_t n = (size_t)K * Nc;
  size_t i = (size_t)blockIdx.x * blockDim.x + threadIdx.x;
  size_t st = (size_t)gridDim.x * blockDim.x;
  for (; i < n; i += st) {
    int k = (int)(i / Nc), c = (int)(i % Nc);
    Wt[(size_t)c * K + k] = (_Float16)W[i];
  }
}

// ---------- WMMA GEMM: C[M,Nc] = Ah[M,K] @ Wt^T + bias ----------
// Ah: f16 row-major [M,K]; Wt: f16 row-major [Nc,K] (W transposed).
// Block: 256 threads = 8 waves; wave computes 16 rows x 64 cols.
// B panel (shared by all 8 waves) is staged into LDS in K-chunks with
// async-to-LDS loads (ASYNCcnt), then WMMA reads B fragments from LDS.
__device__ __forceinline__ void store_tile(float* __restrict__ C, const v8f& a, int row0,
                                           int hi, int col, int M, int Nc, float bv, int relu) {
#pragma unroll
  for (int j = 0; j < 8; ++j) {
    int row = row0 + hi * 8 + j;
    if (row < M) {
      float o = a[j] + bv;
      if (relu) o = fmaxf(o, 0.0f);
      C[(size_t)row * Nc + col] = o;
    }
  }
}

__global__ void __launch_bounds__(256)
gemm_wmma_f16(const _Float16* __restrict__ Ah, const _Float16* __restrict__ Wt,
              const float* __restrict__ bias, float* __restrict__ C,
              int M, int K, int Nc, int relu) {
  extern __shared__ _Float16 Bs[];  // [64][GEMM_KS]
  const int lane = threadIdx.x & 31;
  const int wave = threadIdx.x >> 5;
  const int l15  = lane & 15;
  const int hi   = lane >> 4;  // 0 or 1
  const int row0 = blockIdx.x * 128 + wave * 16;
  const int col0 = blockIdx.y * 64;

  int arow = row0 + l15;
  if (arow >= M) arow = M - 1;  // clamp; masked at store
  const _Float16* Ap = Ah + (size_t)arow * K + hi * 16;

  const unsigned ldsbase = (unsigned)(unsigned long long)(const void*)Bs;
  // LDS read pointers for the 4 column tiles (constant across K-chunks)
  const _Float16* bp0 = Bs + (size_t)(( 0 + l15) * GEMM_KS + hi * 16);
  const _Float16* bp1 = Bs + (size_t)((16 + l15) * GEMM_KS + hi * 16);
  const _Float16* bp2 = Bs + (size_t)((32 + l15) * GEMM_KS + hi * 16);
  const _Float16* bp3 = Bs + (size_t)((48 + l15) * GEMM_KS + hi * 16);

  v8f a0 = {}, a1 = {}, a2 = {}, a3 = {};
  for (int kb = 0; kb < K; kb += GEMM_KC) {
    int kc = K - kb; if (kc > GEMM_KC) kc = GEMM_KC;
    // --- stage B panel chunk: 64 columns x kc halfs, async global -> LDS ---
    {
      const int nch = kc >> 3;  // 16-byte chunks per column
      for (int cc = wave; cc < 64; cc += 8) {
        const _Float16* gcol = Wt + (size_t)(col0 + cc) * K + kb;
        unsigned lcol = ldsbase + (unsigned)(cc * (GEMM_KS * 2));
        for (int ch = lane; ch < nch; ch += 32) {
          const _Float16* gp = gcol + ch * 8;
          unsigned loff = lcol + (unsigned)(ch * 16);
          asm volatile("global_load_async_to_lds_b128 %0, %1, off"
                       :: "v"(loff), "v"(gp) : "memory");
        }
      }
      asm volatile("s_wait_asynccnt 0x0" ::: "memory");
      __syncthreads();
    }
    // --- WMMA over the chunk: A from global, B fragments from LDS ---
#pragma unroll 2
    for (int k = 0; k < kc; k += 32) {
      v8h lo  = *(const v8h*)(Ap + kb + k);
      v8h hh  = *(const v8h*)(Ap + kb + k + 8);
      v16h af = __builtin_shufflevector(lo, hh, 0,1,2,3,4,5,6,7,8,9,10,11,12,13,14,15);
      v8h x0 = *(const v8h*)(bp0 + k), x1 = *(const v8h*)(bp0 + k + 8);
      v8h y0 = *(const v8h*)(bp1 + k), y1 = *(const v8h*)(bp1 + k + 8);
      v8h z0 = *(const v8h*)(bp2 + k), z1 = *(const v8h*)(bp2 + k + 8);
      v8h w0 = *(const v8h*)(bp3 + k), w1 = *(const v8h*)(bp3 + k + 8);
      v16h b0 = __builtin_shufflevector(x0, x1, 0,1,2,3,4,5,6,7,8,9,10,11,12,13,14,15);
      v16h b1 = __builtin_shufflevector(y0, y1, 0,1,2,3,4,5,6,7,8,9,10,11,12,13,14,15);
      v16h b2 = __builtin_shufflevector(z0, z1, 0,1,2,3,4,5,6,7,8,9,10,11,12,13,14,15);
      v16h b3 = __builtin_shufflevector(w0, w1, 0,1,2,3,4,5,6,7,8,9,10,11,12,13,14,15);
      a0 = __builtin_amdgcn_wmma_f32_16x16x32_f16(false, af, false, b0, (short)0, a0, false, false);
      a1 = __builtin_amdgcn_wmma_f32_16x16x32_f16(false, af, false, b1, (short)0, a1, false, false);
      a2 = __builtin_amdgcn_wmma_f32_16x16x32_f16(false, af, false, b2, (short)0, a2, false, false);
      a3 = __builtin_amdgcn_wmma_f32_16x16x32_f16(false, af, false, b3, (short)0, a3, false, false);
    }
    __syncthreads();  // protect Bs before next chunk overwrites it
  }
  {
    int c0 = col0 + l15;
    float bv0 = bias ? bias[c0 +  0] : 0.f;
    float bv1 = bias ? bias[c0 + 16] : 0.f;
    float bv2 = bias ? bias[c0 + 32] : 0.f;
    float bv3 = bias ? bias[c0 + 48] : 0.f;
    store_tile(C, a0, row0, hi, c0 +  0, M, Nc, bv0, relu);
    store_tile(C, a1, row0, hi, c0 + 16, M, Nc, bv1, relu);
    store_tile(C, a2, row0, hi, c0 + 32, M, Nc, bv2, relu);
    store_tile(C, a3, row0, hi, c0 + 48, M, Nc, bv3, relu);
  }
}

// ---------- fused (x [+xb] [+y [+yb]]) -> LayerNorm -> optional relu ----------
__global__ void fused_ln_kernel(const float* __restrict__ x, const float* __restrict__ xb,
                                const float* __restrict__ y, const float* __restrict__ yb,
                                const float* __restrict__ g, const float* __restrict__ b,
                                float* __restrict__ out, int D, int relu) {
  __shared__ float red[512];
  int r = blockIdx.x, c = threadIdx.x;
  float v = x[(size_t)r * D + c];
  if (xb) v += xb[c];
  if (y) { v += y[(size_t)r * D + c]; if (yb) v += yb[c]; }
  red[c] = v; __syncthreads();
  for (int off = D >> 1; off; off >>= 1) { if (c < off) red[c] += red[c + off]; __syncthreads(); }
  float mu = red[0] / (float)D;
  __syncthreads();
  float d = v - mu;
  red[c] = d * d; __syncthreads();
  for (int off = D >> 1; off; off >>= 1) { if (c < off) red[c] += red[c + off]; __syncthreads(); }
  float var = red[0] / (float)D;
  float o = d * rsqrtf(var + 1e-5f) * g[c] + b[c];
  if (relu) o = fmaxf(o, 0.f);
  out[(size_t)r * D + c] = o;
}

// ---------- GAT attention logits: al[n,h] = sum_d xp[n,h*64+d]*a[h*64+d] ----------
__global__ void gat_logits_kernel(const float* __restrict__ xp, const float* __restrict__ as,
                                  const float* __restrict__ ad, float* __restrict__ alS,
                                  float* __restrict__ alD) {
  __shared__ float s1[256], s2[256];
  int n = blockIdx.x, c = threadIdx.x;
  float v = xp[(size_t)n * 256 + c];
  s1[c] = v * as[c]; s2[c] = v * ad[c]; __syncthreads();
  for (int off = 32; off; off >>= 1) {
    if ((c & 63) < off) { s1[c] += s1[c + off]; s2[c] += s2[c + off]; }
    __syncthreads();
  }
  if ((c & 63) == 0) { int h = c >> 6; alS[n * 4 + h] = s1[c]; alD[n * 4 + h] = s2[c]; }
}

// ---------- edge passes (segment softmax over dst + scatter) ----------
__global__ void edge_pass_a(const int* __restrict__ src, const int* __restrict__ dst,
                            const float* __restrict__ alS, const float* __restrict__ alD,
                            float* __restrict__ elog, unsigned* __restrict__ dmax, int E) {
  int i = blockIdx.x * blockDim.x + threadIdx.x;
  if (i >= E * 4) return;
  int e = i >> 2, h = i & 3;
  float l = alS[src[e] * 4 + h] + alD[dst[e] * 4 + h];
  l = (l > 0.f) ? l : 0.2f * l;  // leaky_relu 0.2
  elog[i] = l;
  atomicMax(&dmax[dst[e] * 4 + h], encF(l));
}
__global__ void edge_pass_b(const int* __restrict__ dst, float* __restrict__ elog,
                            const unsigned* __restrict__ dmax, float* __restrict__ dsum, int E) {
  int i = blockIdx.x * blockDim.x + threadIdx.x;
  if (i >= E * 4) return;
  int e = i >> 2, h = i & 3;
  int d = dst[e];
  float ex = expf(elog[i] - decF(dmax[d * 4 + h]));
  elog[i] = ex;
  atomicAdd(&dsum[d * 4 + h], ex);
}
// 4 edges per block; prefetch next edge's feature row (global_prefetch) while
// scattering the current one.
__global__ void edge_pass_c(const int* __restrict__ src, const int* __restrict__ dst,
                            const float* __restrict__ xp, const float* __restrict__ ex,
                            const float* __restrict__ dsum, float* __restrict__ acc, int E) {
  int c = threadIdx.x;
  int e0 = blockIdx.x * 4;
#pragma unroll
  for (int i = 0; i < 4; ++i) {
    int e = e0 + i;
    if (e >= E) return;
    if (i < 3 && e + 1 < E)
      __builtin_prefetch(&xp[(size_t)src[e + 1] * 256 + c], 0, 1);
    int s = src[e], d = dst[e];
    int h = c >> 6;
    float alpha = ex[e * 4 + h] / (dsum[d * 4 + h] + 1e-16f);
    atomicAdd(&acc[(size_t)d * 256 + c], xp[(size_t)s * 256 + c] * alpha);
  }
}

// ---------- per-node dot with small weight vector (+bias, optional sigmoid) ----------
__global__ void node_dot_kernel(const float* __restrict__ H, const float* __restrict__ w,
                                const float* __restrict__ b, float* __restrict__ out,
                                int D, int sigm) {
  __shared__ float red[128];
  int n = blockIdx.x, c = threadIdx.x;
  red[c] = H[(size_t)n * D + c] * w[c]; __syncthreads();
  for (int off = D >> 1; off; off >>= 1) { if (c < off) red[c] += red[c + off]; __syncthreads(); }
  if (c == 0) {
    float o = red[0] + b[0];
    if (sigm) o = 1.f / (1.f + expf(-o));
    out[n] = o;
  }
}

// ---------- per-graph softmax over nodes ----------
__global__ void seg_max_1d(const float* __restrict__ v, const int* __restrict__ seg,
                           unsigned* __restrict__ m, int N) {
  int n = blockIdx.x * blockDim.x + threadIdx.x;
  if (n < N) atomicMax(&m[seg[n]], encF(v[n]));
}
__global__ void seg_exp_sum_1d(float* __restrict__ v, const int* __restrict__ seg,
                               const unsigned* __restrict__ m, float* __restrict__ s, int N) {
  int n = blockIdx.x * blockDim.x + threadIdx.x;
  if (n < N) {
    float e = expf(v[n] - decF(m[seg[n]]));
    v[n] = e;
    atomicAdd(&s[seg[n]], e);
  }
}

// ---------- pooling accumulation ----------
__global__ void pool_accum(const float* __restrict__ node, const int* __restrict__ batch,
                           const float* __restrict__ exA, const float* __restrict__ gsum,
                           const float* __restrict__ wgt, float* __restrict__ meanA,
                           unsigned* __restrict__ maxA, float* __restrict__ attA,
                           float* __restrict__ wA, float* __restrict__ wsum,
                           float* __restrict__ cnt) {
  int n = blockIdx.x, c = threadIdx.x;
  int g = batch[n];
  float v = node[(size_t)n * 256 + c];
  atomicAdd(&meanA[(size_t)g * 256 + c], v);
  atomicMax(&maxA[(size_t)g * 256 + c], encF(v));
  float aw = exA[n] / (gsum[g] + 1e-16f);
  atomicAdd(&attA[(size_t)g * 256 + c], v * aw);
  float w = wgt[n];
  atomicAdd(&wA[(size_t)g * 256 + c], v * w);
  if (c == 0) { atomicAdd(&wsum[g], w); atomicAdd(&cnt[g], 1.0f); }
}

__global__ void finalize_ge(const float* __restrict__ meanA, const unsigned* __restrict__ maxA,
                            const float* __restrict__ attA, const float* __restrict__ wA,
                            const float* __restrict__ wsum, const float* __restrict__ cnt,
                            float* __restrict__ ge) {
  int g = blockIdx.x, c = threadIdx.x;  // blockDim = 1024
  float o;
  if (c < 256)      o = meanA[(size_t)g * 256 + c] / cnt[g];
  else if (c < 512) o = decF(maxA[(size_t)g * 256 + (c - 256)]);
  else if (c < 768) o = attA[(size_t)g * 256 + (c - 512)];
  else              o = wA[(size_t)g * 256 + (c - 768)] / (wsum[g] + 1e-8f);
  ge[(size_t)g * 1024 + c] = o;
}

// ---------- final tiny GEMM [G,256] @ [256,2] ----------
__global__ void head_out(const float* __restrict__ h2, const float* __restrict__ w3,
                         const float* __restrict__ b3, float* __restrict__ out, int G) {
  int i = blockIdx.x * blockDim.x + threadIdx.x;
  if (i >= G * 2) return;
  int g = i >> 1, cls = i & 1;
  float s = b3[cls];
  for (int k = 0; k < 256; ++k) s += h2[(size_t)g * 256 + k] * w3[k * 2 + cls];
  out[i] = s;
}

// ============================================================================
extern "C" void kernel_launch(void* const* d_in, const int* in_sizes, int n_in,
                              void* d_out, int out_size, void* d_ws, size_t ws_size,
                              hipStream_t stream) {
  (void)n_in; (void)ws_size;
  const float* x      = (const float*)d_in[0];
  const int*   eidx   = (const int*)d_in[1];
  const int*   batch  = (const int*)d_in[2];
  const float* w_enc  = (const float*)d_in[3];
  const float* b_enc  = (const float*)d_in[4];
  const float* ln_g   = (const float*)d_in[5];
  const float* ln_b   = (const float*)d_in[6];
  const float* gat0_w = (const float*)d_in[7];
  const float* gat0_as= (const float*)d_in[8];
  const float* gat0_ad= (const float*)d_in[9];
  const float* gat0_b = (const float*)d_in[10];
  const float* gat1_w = (const float*)d_in[11];
  const float* gat1_as= (const float*)d_in[12];
  const float* gat1_ad= (const float*)d_in[13];
  const float* gat1_b = (const float*)d_in[14];
  const float* res_w  = (const float*)d_in[15];
  const float* res_b  = (const float*)d_in[16];
  const float* gn_g   = (const float*)d_in[17];
  const float* gn_b   = (const float*)d_in[18];
  const float* ap_w1  = (const float*)d_in[19];
  const float* ap_b1  = (const float*)d_in[20];
  const float* ap_w2  = (const float*)d_in[21];
  const float* ap_b2  = (const float*)d_in[22];
  const float* wp_w1  = (const float*)d_in[23];
  const float* wp_b1  = (const float*)d_in[24];
  const float* wp_w2  = (const float*)d_in[25];
  const float* wp_b2  = (const float*)d_in[26];
  const float* h_w1   = (const float*)d_in[27];
  const float* h_b1   = (const float*)d_in[28];
  const float* h_ln1g = (const float*)d_in[29];
  const float* h_ln1b = (const float*)d_in[30];
  const float* h_w2   = (const float*)d_in[31];
  const float* h_b2   = (const float*)d_in[32];
  const float* h_ln2g = (const float*)d_in[33];
  const float* h_ln2b = (const float*)d_in[34];
  const float* h_w3   = (const float*)d_in[35];
  const float* h_b3   = (const float*)d_in[36];

  const int N = in_sizes[0] / 128;
  const int E = in_sizes[1] / 2;
  const int G = out_size / 2;
  const int* src = eidx;
  const int* dst = eidx + E;

  // ---- workspace carve (256B aligned) ----
  char* wp = (char*)d_ws;
  auto alloc = [&](size_t bytes) -> void* {
    void* p = (void*)wp; wp += (bytes + 255) & ~(size_t)255; return p;
  };
  float*    P0   = (float*)alloc((size_t)N * 256 * 4);   // node_raw -> acc0 -> acc1
  float*    P1   = (float*)alloc((size_t)N * 256 * 4);   // node_ln -> resp -> node1 -> node2
  float*    P2   = (float*)alloc((size_t)N * 256 * 4);   // xp0 -> xp1 -> pool hiddens
  _Float16* AH   = (_Float16*)alloc((size_t)N * 256 * 2);
  _Float16* WT   = (_Float16*)alloc((size_t)1024 * 512 * 2);
  float*    elog = (float*)alloc((size_t)E * 4 * 4);
  unsigned* dmax = (unsigned*)alloc((size_t)N * 4 * 4);
  float*    dsum = (float*)alloc((size_t)N * 4 * 4);
  float*    alS  = (float*)alloc((size_t)N * 4 * 4);
  float*    alD  = (float*)alloc((size_t)N * 4 * 4);
  float*    att  = (float*)alloc((size_t)N * 4);
  float*    wgt  = (float*)alloc((size_t)N * 4);
  unsigned* gmax = (unsigned*)alloc((size_t)G * 4);
  float*    gsum = (float*)alloc((size_t)G * 4);
  float*    meanA= (float*)alloc((size_t)G * 256 * 4);
  unsigned* maxA = (unsigned*)alloc((size_t)G * 256 * 4);
  float*    attA = (float*)alloc((size_t)G * 256 * 4);
  float*    wA   = (float*)alloc((size_t)G * 256 * 4);
  float*    wsum = (float*)alloc((size_t)G * 4);
  float*    cnt  = (float*)alloc((size_t)G * 4);
  float*    ge   = (float*)alloc((size_t)G * 1024 * 4);
  float*    h1t  = (float*)alloc((size_t)G * 512 * 4);
  float*    h1   = (float*)alloc((size_t)G * 512 * 4);
  float*    h2t  = (float*)alloc((size_t)G * 256 * 4);
  float*    h2   = (float*)alloc((size_t)G * 256 * 4);

  auto cdiv = [](int a, int b) { return (a + b - 1) / b; };

  // convert A (f32->f16) and W (f32->f16 transposed), then WMMA GEMM
  auto gemm = [&](const float* Af32, size_t an, const float* Wf32, int K, int Nc,
                  const float* bias, float* C, int M, int relu) {
    f2h_kernel<<<2048, 256, 0, stream>>>(Af32, AH, an);
    f2h_tr_kernel<<<2048, 256, 0, stream>>>(Wf32, WT, K, Nc);
    dim3 g(cdiv(M, 128), Nc / 64);
    gemm_wmma_f16<<<g, 256, GEMM_SHM, stream>>>(AH, WT, bias, C, M, K, Nc, relu);
  };

  const int EB = cdiv(E * 4, 256);
  const int NB = cdiv(N, 256);
  const int CB = cdiv(E, 4);

  // --- encoder: node_raw = x @ w_enc + b_enc ; node_ln = LN(node_raw)
  gemm(x, (size_t)N * 128, w_enc, 128, 256, b_enc, P0, N, 0);
  fused_ln_kernel<<<N, 256, 0, stream>>>(P0, nullptr, nullptr, nullptr, ln_g, ln_b, P1, 256, 0);

  // --- GAT layer 0 ---
  gemm(P1, (size_t)N * 256, gat0_w, 256, 256, nullptr, P2, N, 0);       // xp0
  gemm(P0, (size_t)N * 256, res_w, 256, 256, res_b, P1, N, 0);          // resp
  gat_logits_kernel<<<N, 256, 0, stream>>>(P2, gat0_as, gat0_ad, alS, alD);
  fill_u32<<<1024, 256, 0, stream>>>(dmax, 0u, (size_t)N * 4);
  fill_f32<<<1024, 256, 0, stream>>>(dsum, 0.f, (size_t)N * 4);
  fill_f32<<<4096, 256, 0, stream>>>(P0, 0.f, (size_t)N * 256);         // acc0
  edge_pass_a<<<EB, 256, 0, stream>>>(src, dst, alS, alD, elog, dmax, E);
  edge_pass_b<<<EB, 256, 0, stream>>>(dst, elog, dmax, dsum, E);
  edge_pass_c<<<CB, 256, 0, stream>>>(src, dst, P2, elog, dsum, P0, E);
  // node1 = relu(LN(acc0 + gat0_b + resp))   (resp already has res_b)
  fused_ln_kernel<<<N, 256, 0, stream>>>(P0, gat0_b, P1, nullptr, gn_g, gn_b, P1, 256, 1);

  // --- GAT layer 1 (identity residual) ---
  gemm(P1, (size_t)N * 256, gat1_w, 256, 256, nullptr, P2, N, 0);       // xp1
  gat_logits_kernel<<<N, 256, 0, stream>>>(P2, gat1_as, gat1_ad, alS, alD);
  fill_u32<<<1024, 256, 0, stream>>>(dmax, 0u, (size_t)N * 4);
  fill_f32<<<1024, 256, 0, stream>>>(dsum, 0.f, (size_t)N * 4);
  fill_f32<<<4096, 256, 0, stream>>>(P0, 0.f, (size_t)N * 256);         // acc1
  edge_pass_a<<<EB, 256, 0, stream>>>(src, dst, alS, alD, elog, dmax, E);
  edge_pass_b<<<EB, 256, 0, stream>>>(dst, elog, dmax, dsum, E);
  edge_pass_c<<<CB, 256, 0, stream>>>(src, dst, P2, elog, dsum, P0, E);
  // node2 = relu(LN(acc1 + gat1_b + node1))
  fused_ln_kernel<<<N, 256, 0, stream>>>(P0, gat1_b, P1, nullptr, gn_g, gn_b, P1, 256, 1);

  // --- attention / weight pooling scores ---
  gemm(P1, (size_t)N * 256, ap_w1, 256, 128, ap_b1, P2, N, 1);          // relu(node2@ap_w1+b1)
  node_dot_kernel<<<N, 128, 0, stream>>>(P2, ap_w2, ap_b2, att, 128, 0);
  gemm(P1, (size_t)N * 256, wp_w1, 256, 64, wp_b1, P2, N, 1);           // relu(node2@wp_w1+b1)
  node_dot_kernel<<<N, 64, 0, stream>>>(P2, wp_w2, wp_b2, wgt, 64, 1);  // sigmoid

  // per-graph softmax of att
  fill_u32<<<64, 256, 0, stream>>>(gmax, 0u, (size_t)G);
  fill_f32<<<64, 256, 0, stream>>>(gsum, 0.f, (size_t)G);
  seg_max_1d<<<NB, 256, 0, stream>>>(att, batch, gmax, N);
  seg_exp_sum_1d<<<NB, 256, 0, stream>>>(att, batch, gmax, gsum, N);    // att := exp(att-max)

  // --- pooling ---
  fill_f32<<<1024, 256, 0, stream>>>(meanA, 0.f, (size_t)G * 256);
  fill_u32<<<1024, 256, 0, stream>>>(maxA, 0u, (size_t)G * 256);
  fill_f32<<<1024, 256, 0, stream>>>(attA, 0.f, (size_t)G * 256);
  fill_f32<<<1024, 256, 0, stream>>>(wA, 0.f, (size_t)G * 256);
  fill_f32<<<64, 256, 0, stream>>>(wsum, 0.f, (size_t)G);
  fill_f32<<<64, 256, 0, stream>>>(cnt, 0.f, (size_t)G);
  pool_accum<<<N, 256, 0, stream>>>(P1, batch, att, gsum, wgt, meanA, maxA, attA, wA, wsum, cnt);
  finalize_ge<<<G, 1024, 0, stream>>>(meanA, maxA, attA, wA, wsum, cnt, ge);

  // --- head MLP ---
  gemm(ge, (size_t)G * 1024, h_w1, 1024, 512, h_b1, h1t, G, 0);
  fused_ln_kernel<<<G, 512, 0, stream>>>(h1t, nullptr, nullptr, nullptr, h_ln1g, h_ln1b, h1, 512, 1);
  gemm(h1, (size_t)G * 512, h_w2, 512, 256, h_b2, h2t, G, 0);
  fused_ln_kernel<<<G, 256, 0, stream>>>(h2t, nullptr, nullptr, nullptr, h_ln2g, h_ln2b, h2, 256, 1);
  head_out<<<cdiv(G * 2, 256), 256, 0, stream>>>(h2, h_w3, h_b3, (float*)d_out, G);
}
